// MPNNet_Atom_4148938408839
// MI455X (gfx1250) — compile-verified, matrix operand order checked by reading, not run on hardware
//
#include <hip/hip_runtime.h>
#include <hip/hip_bf16.h>

typedef __attribute__((ext_vector_type(16))) _Float16 v16h;
typedef __attribute__((ext_vector_type(8)))  float    v8f;

#define N_NODES     25000
#define N_EDGES     100000
#define N_GRAPHS    1000
#define DIM         64
#define NUM_FEAT    14
#define EDGE_FEAT   4
#define HIDDEN_EDGE 128
#define WMAT        (DIM * DIM)      /* 4096 */
#define CHUNK       10000            /* edges per chunk; 10 chunks; 10000/16 = 625 m-tiles */
#define GN          256              /* N columns staged in LDS per block (64 KB) */
#define MIT         5                /* m-tiles per block; 625 = 125 * 5 */
#define NEG_SLOPE   0.01f

__device__ __forceinline__ float lrelu(float v) { return v > 0.f ? v : NEG_SLOPE * v; }
__device__ __forceinline__ float sigm(float v)  { return 1.f / (1.f + __expf(-v)); }

__device__ __forceinline__ void atomicMaxFloat(float* addr, float val) {
    // IEEE ordering trick: non-negative -> int max, negative -> uint min
    if (val >= 0.f) atomicMax((int*)addr, __float_as_int(val));
    else            atomicMin((unsigned int*)addr, __float_as_uint(val));
}

union V16HU { uint4 q[2]; v16h v; };
union W8HU  { uint4 q; _Float16 f[8]; };

// ---------------------------------------------------------------------------
// lin0: h = leaky_relu(x @ lin0_w.T + b)   [25000,14] x [64,14]^T -> [25000,64]
// ---------------------------------------------------------------------------
__global__ __launch_bounds__(256) void lin0_kernel(
    const float* __restrict__ x, const float* __restrict__ w,
    const float* __restrict__ b, float* __restrict__ h)
{
    int ln = threadIdx.x >> 6;            // node within block (0..3)
    int o  = threadIdx.x & 63;
    int n  = blockIdx.x * 4 + ln;         // 6250 blocks, exact
    float acc = b[o];
#pragma unroll
    for (int f = 0; f < NUM_FEAT; ++f)
        acc += x[n * NUM_FEAT + f] * w[o * NUM_FEAT + f];
    h[n * DIM + o] = lrelu(acc);
}

// ---------------------------------------------------------------------------
// edge MLP layer 1: h1 = leaky_relu(edge_attr @ w1.T + b1) -> f16 [E,128]
// ---------------------------------------------------------------------------
__global__ __launch_bounds__(256) void edge_mlp1_kernel(
    const float* __restrict__ ea, const float* __restrict__ w1,
    const float* __restrict__ b1, _Float16* __restrict__ h1)
{
    int idx = blockIdx.x * 256 + threadIdx.x;     // e*128 + j
    if (idx >= N_EDGES * HIDDEN_EDGE) return;
    int e = idx >> 7, j = idx & 127;
    float acc = b1[j];
#pragma unroll
    for (int f = 0; f < EDGE_FEAT; ++f)
        acc += ea[e * EDGE_FEAT + f] * w1[j * EDGE_FEAT + f];
    h1[idx] = (_Float16)lrelu(acc);
}

// ---------------------------------------------------------------------------
// net_w2 fp32 -> f16 with N-permutation so the GEMM emits W_e in transposed
// per-edge layout [e][o][d] (o = out dim, d = in dim): n' = o*64+d maps to
// original n = d*64+o. Makes edge_msg loads contiguous b128.
// ---------------------------------------------------------------------------
__global__ __launch_bounds__(256) void w2_convert_kernel(
    const float* __restrict__ w2, _Float16* __restrict__ w2h)
{
    int i = blockIdx.x * 256 + threadIdx.x;       // dest index n'*128 + k
    if (i >= WMAT * HIDDEN_EDGE) return;
    int np = i >> 7, k = i & 127;
    int n  = ((np & 63) << 6) | (np >> 6);        // inverse permute
    w2h[i] = (_Float16)w2[n * HIDDEN_EDGE + k];
}

// ---------------------------------------------------------------------------
// W_e chunk GEMM (WMMA), B staged in LDS:
//   Wc[m, n'] = f16( sum_k h1[ebase+m,k] * w2h[n'*128+k] + b2[perm(n')] )
// block = 256 thr = 8 waves; LDS holds B slab GN=256 cols x 128 k (64 KB).
// wave -> 2 n-tiles (32 cols); block loops MIT=5 m-tiles reusing the slab.
// grid = (4096/GN, 625/MIT) = (16, 125)
// ---------------------------------------------------------------------------
__global__ __launch_bounds__(256) void we_gemm_kernel(
    const _Float16* __restrict__ h1,   // [N_EDGES,128]
    const _Float16* __restrict__ w2h,  // [4096,128] permuted
    const float*    __restrict__ b2,   // [4096] original order
    _Float16*       __restrict__ wout, // [CHUNK,4096] in [m][o][d] layout
    int ebase)
{
    __shared__ __align__(16) _Float16 Bsh[GN * HIDDEN_EDGE];  // 64 KB

    const int tid  = threadIdx.x;
    const int lane = tid & 31;
    const int wave = tid >> 5;
    const int half = lane >> 4;        // k-half select per ISA 16-bit layout
    const int l16  = lane & 15;
    const int nbase = blockIdx.x * GN;

    // cooperative stage of B slab: GN rows x 256 B = 4096 uint4
    {
        const uint4* src = (const uint4*)(w2h + (size_t)nbase * HIDDEN_EDGE);
        uint4*       dst = (uint4*)Bsh;
#pragma unroll
        for (int i = 0; i < (GN * HIDDEN_EDGE) / 8 / 256; ++i)
            dst[tid + i * 256] = src[tid + i * 256];
    }
    __syncthreads();

    const int nw = wave * 32;          // wave's n-offset within slab

    for (int mi = 0; mi < MIT; ++mi) {
        const int m0 = (blockIdx.y * MIT + mi) * 16;
        const _Float16* aptr =
            h1 + (size_t)(ebase + m0 + l16) * HIDDEN_EDGE + half * 8;

        v8f c0 = {}, c1 = {};
#pragma unroll
        for (int k0 = 0; k0 < HIDDEN_EDGE; k0 += 32) {
            V16HU a;
            a.q[0] = *(const uint4*)(aptr + k0);        // k = k0 + half*8 + 0..7
            a.q[1] = *(const uint4*)(aptr + k0 + 16);   // k = k0+16 + half*8 + 0..7
            const _Float16* bs = Bsh + (nw + l16) * HIDDEN_EDGE + k0 + half * 8;
            V16HU b0, b1v;
            b0.q[0]  = *(const uint4*)(bs);
            b0.q[1]  = *(const uint4*)(bs + 16);
            b1v.q[0] = *(const uint4*)(bs + 16 * HIDDEN_EDGE);
            b1v.q[1] = *(const uint4*)(bs + 16 * HIDDEN_EDGE + 16);
            c0 = __builtin_amdgcn_wmma_f32_16x16x32_f16(false, a.v, false, b0.v,
                                                        (short)0, c0, false, false);
            c1 = __builtin_amdgcn_wmma_f32_16x16x32_f16(false, a.v, false, b1v.v,
                                                        (short)0, c1, false, false);
        }
        // store: lane -> n' = nbase + nw + t*16 + l16 ; vgpr r -> m = m0 + r + 8*half
#pragma unroll
        for (int t = 0; t < 2; ++t) {
            const int np   = nbase + nw + t * 16 + l16;
            const int norg = ((np & 63) << 6) | (np >> 6);
            const float bias = b2[norg];
            const v8f  c = t ? c1 : c0;
#pragma unroll
            for (int r = 0; r < 8; ++r) {
                int m = m0 + r + half * 8;
                wout[(size_t)m * WMAT + np] = (_Float16)(c[r] + bias);
            }
        }
    }
}

// ---------------------------------------------------------------------------
// per-edge matvec + scatter: aggr[dst] += out[src] @ W_e
// W in [e][o][d] layout -> each thread reads its 64 d-values as 8 x b128
// block 256 = 4 edges x 64 output lanes
// ---------------------------------------------------------------------------
__global__ __launch_bounds__(256) void edge_msg_kernel(
    const _Float16* __restrict__ wchunk, const float* __restrict__ hcur,
    const int* __restrict__ ei, float* __restrict__ aggr, int ebase)
{
    __shared__ float vsh[4][DIM];
    int le = threadIdx.x >> 6;
    int o  = threadIdx.x & 63;
    int e  = blockIdx.x * 4 + le;        // CHUNK/4 blocks, exact
    int ge = ebase + e;
    int src = ei[ge];
    vsh[le][o] = hcur[src * DIM + o];
    __syncthreads();

    const _Float16* w = wchunk + (size_t)e * WMAT + o * DIM;
    float acc = 0.f;
#pragma unroll
    for (int d8 = 0; d8 < DIM; d8 += 8) {
        W8HU wv;
        wv.q = *(const uint4*)(w + d8);
#pragma unroll
        for (int j = 0; j < 8; ++j)
            acc += vsh[le][d8 + j] * (float)wv.f[j];
    }
    int dst = ei[N_EDGES + ge];
    atomicAdd(&aggr[dst * DIM + o], acc);
}

// ---------------------------------------------------------------------------
// degree
// ---------------------------------------------------------------------------
__global__ __launch_bounds__(256) void deg_kernel(const int* __restrict__ ei,
                                                  float* __restrict__ invdeg)
{
    int e = blockIdx.x * 256 + threadIdx.x;
    if (e < N_EDGES) atomicAdd(&invdeg[ei[N_EDGES + e]], 1.0f);
}
__global__ __launch_bounds__(256) void invdeg_kernel(float* __restrict__ invdeg)
{
    int n = blockIdx.x * 256 + threadIdx.x;
    if (n < N_NODES) { float d = invdeg[n]; invdeg[n] = d > 0.f ? 1.f / d : 0.f; }
}

// ---------------------------------------------------------------------------
// scatter-mean + root + bias + leaky_relu, then one GRU step, in-place on h
// block 256 = 4 nodes x 64 dims
// ---------------------------------------------------------------------------
__global__ __launch_bounds__(256) void node_update_kernel(
    const float* __restrict__ aggr, const float* __restrict__ invdeg,
    const float* __restrict__ conv_root, const float* __restrict__ conv_bias,
    const float* __restrict__ w_ih, const float* __restrict__ w_hh,
    const float* __restrict__ b_ih, const float* __restrict__ b_hh,
    float* __restrict__ h)
{
    __shared__ float hsh[4][DIM];
    __shared__ float msh[4][DIM];
    int ln = threadIdx.x >> 6;
    int o  = threadIdx.x & 63;
    int n  = blockIdx.x * 4 + ln;       // 6250 blocks, exact

    hsh[ln][o] = h[n * DIM + o];
    __syncthreads();

    float root = 0.f;
#pragma unroll 16
    for (int d = 0; d < DIM; ++d) root += hsh[ln][d] * conv_root[d * DIM + o];
    float mv = lrelu(aggr[n * DIM + o] * invdeg[n] + root + conv_bias[o]);
    msh[ln][o] = mv;
    __syncthreads();

    float gi_r = b_ih[o], gi_z = b_ih[64 + o], gi_n = b_ih[128 + o];
    float gh_r = b_hh[o], gh_z = b_hh[64 + o], gh_n = b_hh[128 + o];
#pragma unroll 8
    for (int d = 0; d < DIM; ++d) {
        float md = msh[ln][d], hd = hsh[ln][d];
        gi_r += md * w_ih[o * DIM + d];
        gi_z += md * w_ih[(64 + o) * DIM + d];
        gi_n += md * w_ih[(128 + o) * DIM + d];
        gh_r += hd * w_hh[o * DIM + d];
        gh_z += hd * w_hh[(64 + o) * DIM + d];
        gh_n += hd * w_hh[(128 + o) * DIM + d];
    }
    float r  = sigm(gi_r + gh_r);
    float z  = sigm(gi_z + gh_z);
    float nn = tanhf(gi_n + r * gh_n);
    h[n * DIM + o] = (1.f - z) * nn + z * hsh[ln][o];
}

// ---------------------------------------------------------------------------
// Set2Set
// ---------------------------------------------------------------------------
__global__ __launch_bounds__(64) void lstm_kernel(
    const float* __restrict__ qstar, float* __restrict__ hl, float* __restrict__ cl,
    const float* __restrict__ w_ih, const float* __restrict__ w_hh,
    const float* __restrict__ b_ih, const float* __restrict__ b_hh)
{
    __shared__ float qs[2 * DIM];
    __shared__ float hs[DIM];
    int b = blockIdx.x, j = threadIdx.x;
    qs[j]      = qstar[b * 2 * DIM + j];
    qs[64 + j] = qstar[b * 2 * DIM + 64 + j];
    hs[j]      = hl[b * DIM + j];
    __syncthreads();

    float gi = b_ih[j]       + b_hh[j];
    float gf = b_ih[64 + j]  + b_hh[64 + j];
    float gg = b_ih[128 + j] + b_hh[128 + j];
    float go = b_ih[192 + j] + b_hh[192 + j];
#pragma unroll 8
    for (int k = 0; k < 2 * DIM; ++k) {
        float q = qs[k];
        gi += q * w_ih[j * 128 + k];
        gf += q * w_ih[(64 + j) * 128 + k];
        gg += q * w_ih[(128 + j) * 128 + k];
        go += q * w_ih[(192 + j) * 128 + k];
    }
#pragma unroll 8
    for (int k = 0; k < DIM; ++k) {
        float hh = hs[k];
        gi += hh * w_hh[j * 64 + k];
        gf += hh * w_hh[(64 + j) * 64 + k];
        gg += hh * w_hh[(128 + j) * 64 + k];
        go += hh * w_hh[(192 + j) * 64 + k];
    }
    float c = sigm(gf) * cl[b * DIM + j] + sigm(gi) * tanhf(gg);
    cl[b * DIM + j] = c;
    hl[b * DIM + j] = sigm(go) * tanhf(c);
}

__global__ __launch_bounds__(256) void seg_init_kernel(
    float* __restrict__ emax, float* __restrict__ denom, float* __restrict__ rread)
{
    int i = blockIdx.x * 256 + threadIdx.x;
    if (i < N_GRAPHS) { emax[i] = __int_as_float(0xFF800000); denom[i] = 0.f; }
    if (i < N_GRAPHS * DIM) rread[i] = 0.f;
}

__global__ __launch_bounds__(256) void att_e_kernel(
    const float* __restrict__ h, const float* __restrict__ hl,
    const int* __restrict__ batch, float* __restrict__ ev, float* __restrict__ emax)
{
    __shared__ float red[256];
    int ln = threadIdx.x >> 6, o = threadIdx.x & 63;
    int n = blockIdx.x * 4 + ln;
    int b = batch[n];
    red[threadIdx.x] = h[n * DIM + o] * hl[b * DIM + o];
    __syncthreads();
    for (int s = 32; s >= 1; s >>= 1) {
        if (o < s) red[threadIdx.x] += red[threadIdx.x + s];
        __syncthreads();
    }
    if (o == 0) {
        float e = red[ln * 64];
        ev[n] = e;
        atomicMaxFloat(&emax[b], e);
    }
}

__global__ __launch_bounds__(256) void att_soft_kernel(
    const int* __restrict__ batch, float* __restrict__ ev,
    const float* __restrict__ emax, float* __restrict__ denom)
{
    int n = blockIdx.x * 256 + threadIdx.x;
    if (n >= N_NODES) return;
    int b = batch[n];
    float t = __expf(ev[n] - emax[b]);
    ev[n] = t;
    atomicAdd(&denom[b], t);
}

__global__ __launch_bounds__(256) void att_read_kernel(
    const float* __restrict__ h, const int* __restrict__ batch,
    const float* __restrict__ ev, const float* __restrict__ denom,
    float* __restrict__ rread)
{
    int ln = threadIdx.x >> 6, o = threadIdx.x & 63;
    int n = blockIdx.x * 4 + ln;
    int b = batch[n];
    float a = ev[n] / denom[b];
    atomicAdd(&rread[b * DIM + o], a * h[n * DIM + o]);
}

__global__ __launch_bounds__(256) void qstar_kernel(
    const float* __restrict__ hl, const float* __restrict__ rread,
    float* __restrict__ qstar)
{
    int i = blockIdx.x * 256 + threadIdx.x;
    if (i >= N_GRAPHS * 2 * DIM) return;
    int b = i >> 7, k = i & 127;
    qstar[i] = (k < DIM) ? hl[b * DIM + k] : rread[b * DIM + (k - DIM)];
}

__global__ __launch_bounds__(256) void final_kernel(
    const float* __restrict__ qstar, const float* __restrict__ lw,
    const float* __restrict__ lb, float* __restrict__ out)
{
    int b = blockIdx.x * 256 + threadIdx.x;
    if (b >= N_GRAPHS) return;
    float acc = lb[0];
#pragma unroll 8
    for (int k = 0; k < 2 * DIM; ++k) acc += qstar[b * 2 * DIM + k] * lw[k];
    out[b] = acc;
}

// ---------------------------------------------------------------------------
extern "C" void kernel_launch(void* const* d_in, const int* in_sizes, int n_in,
                              void* d_out, int out_size, void* d_ws, size_t ws_size,
                              hipStream_t stream)
{
    const float* x         = (const float*)d_in[0];
    const int*   ei        = (const int*)  d_in[1];
    const float* ea        = (const float*)d_in[2];
    const int*   batch     = (const int*)  d_in[3];
    const float* lin0_w    = (const float*)d_in[4];
    const float* lin0_b    = (const float*)d_in[5];
    const float* net_w1    = (const float*)d_in[6];
    const float* net_b1    = (const float*)d_in[7];
    const float* net_w2    = (const float*)d_in[8];
    const float* net_b2    = (const float*)d_in[9];
    const float* conv_root = (const float*)d_in[10];
    const float* conv_bias = (const float*)d_in[11];
    const float* gw_ih     = (const float*)d_in[12];
    const float* gw_hh     = (const float*)d_in[13];
    const float* gb_ih     = (const float*)d_in[14];
    const float* gb_hh     = (const float*)d_in[15];
    const float* lw_ih     = (const float*)d_in[16];
    const float* lw_hh     = (const float*)d_in[17];
    const float* lb_ih     = (const float*)d_in[18];
    const float* lb_hh     = (const float*)d_in[19];
    const float* lin_w     = (const float*)d_in[20];
    const float* lin_b     = (const float*)d_in[21];

    // workspace carve (256B aligned slices)
    char* p = (char*)d_ws;
    auto carve = [&](size_t bytes) -> char* {
        char* r = p;
        p += (bytes + 255) & ~(size_t)255;
        return r;
    };
    _Float16* h1     = (_Float16*)carve((size_t)N_EDGES * HIDDEN_EDGE * 2);  // 25.6 MB
    _Float16* w2h    = (_Float16*)carve((size_t)WMAT * HIDDEN_EDGE * 2);     //  1.0 MB
    _Float16* wchunk = (_Float16*)carve((size_t)CHUNK * WMAT * 2);           // 81.9 MB (L2-hot)
    float* h      = (float*)carve((size_t)N_NODES * DIM * 4);                //  6.4 MB
    float* aggr   = (float*)carve((size_t)N_NODES * DIM * 4);                //  6.4 MB
    float* invdeg = (float*)carve((size_t)N_NODES * 4);
    float* qstar  = (float*)carve((size_t)N_GRAPHS * 2 * DIM * 4);
    float* hl     = (float*)carve((size_t)N_GRAPHS * DIM * 4);
    float* cl     = (float*)carve((size_t)N_GRAPHS * DIM * 4);
    float* ev     = (float*)carve((size_t)N_NODES * 4);
    float* emax   = (float*)carve((size_t)N_GRAPHS * 4);
    float* denom  = (float*)carve((size_t)N_GRAPHS * 4);
    float* rread  = (float*)carve((size_t)N_GRAPHS * DIM * 4);
    float* y      = (float*)d_out;

    // --- encoders ---
    lin0_kernel<<<N_NODES / 4, 256, 0, stream>>>(x, lin0_w, lin0_b, h);
    edge_mlp1_kernel<<<(N_EDGES * HIDDEN_EDGE + 255) / 256, 256, 0, stream>>>(ea, net_w1, net_b1, h1);
    w2_convert_kernel<<<(WMAT * HIDDEN_EDGE + 255) / 256, 256, 0, stream>>>(net_w2, w2h);

    hipMemsetAsync(invdeg, 0, (size_t)N_NODES * 4, stream);
    deg_kernel<<<(N_EDGES + 255) / 256, 256, 0, stream>>>(ei, invdeg);
    invdeg_kernel<<<(N_NODES + 255) / 256, 256, 0, stream>>>(invdeg);

    // --- 3 conv + GRU steps; W_e recomputed per step in L2-resident f16 chunks ---
    const int NCHUNK = N_EDGES / CHUNK;  // 10
    for (int step = 0; step < 3; ++step) {
        hipMemsetAsync(aggr, 0, (size_t)N_NODES * DIM * 4, stream);
        for (int c = 0; c < NCHUNK; ++c) {
            we_gemm_kernel<<<dim3(WMAT / GN, (CHUNK / 16) / MIT), 256, 0, stream>>>(
                h1, w2h, net_b2, wchunk, c * CHUNK);
            edge_msg_kernel<<<CHUNK / 4, 256, 0, stream>>>(wchunk, h, ei, aggr, c * CHUNK);
        }
        node_update_kernel<<<N_NODES / 4, 256, 0, stream>>>(
            aggr, invdeg, conv_root, conv_bias, gw_ih, gw_hh, gb_ih, gb_hh, h);
    }

    // --- Set2Set pooling ---
    hipMemsetAsync(qstar, 0, (size_t)N_GRAPHS * 2 * DIM * 4, stream);
    hipMemsetAsync(hl, 0, (size_t)N_GRAPHS * DIM * 4, stream);
    hipMemsetAsync(cl, 0, (size_t)N_GRAPHS * DIM * 4, stream);
    for (int it = 0; it < 3; ++it) {
        lstm_kernel<<<N_GRAPHS, 64, 0, stream>>>(qstar, hl, cl, lw_ih, lw_hh, lb_ih, lb_hh);
        seg_init_kernel<<<(N_GRAPHS * DIM + 255) / 256, 256, 0, stream>>>(emax, denom, rread);
        att_e_kernel<<<N_NODES / 4, 256, 0, stream>>>(h, hl, batch, ev, emax);
        att_soft_kernel<<<(N_NODES + 255) / 256, 256, 0, stream>>>(batch, ev, emax, denom);
        att_read_kernel<<<N_NODES / 4, 256, 0, stream>>>(h, batch, ev, denom, rread);
        qstar_kernel<<<(N_GRAPHS * 2 * DIM + 255) / 256, 256, 0, stream>>>(hl, rread, qstar);
    }
    final_kernel<<<(N_GRAPHS + 255) / 256, 256, 0, stream>>>(qstar, lin_w, lin_b, y);
}
